// SimpleMamba2Like_86947317941000
// MI455X (gfx1250) — compile-verified, matrix operand order checked by reading, not run on hardware
//
#include <hip/hip_runtime.h>
#include <hip/hip_bf16.h>

// ---------------------------------------------------------------------------
// Problem constants (from the reference)
// ---------------------------------------------------------------------------
#define D_MODEL 1024
#define D_INNER 2048
#define D_STATE 16
#define D_CONVK 4
#define DT_RANK 128
#define BATCH   4
#define SEQL    2048
#define MROWS   (BATCH * SEQL)          // 8192 rows for every GEMM

typedef __attribute__((ext_vector_type(16))) __bf16 v16bf;
typedef __attribute__((ext_vector_type(8)))  __bf16 v8bf;
typedef __attribute__((ext_vector_type(4)))  __bf16 v4bf;
typedef __attribute__((ext_vector_type(8)))  float  v8f;

#define EPI_NONE          0
#define EPI_BIAS          1
#define EPI_BIAS_SOFTPLUS 2
#define EPI_SIGMUL        3

__device__ __forceinline__ float sigmoidf_(float x) {
    return 1.0f / (1.0f + __expf(-x));
}
__device__ __forceinline__ float softplusf_(float x) {
    return (x > 20.0f) ? x : log1pf(__expf(x));
}
__device__ __forceinline__ void split_bf16(float x, __bf16& hi, __bf16& lo) {
    hi = (__bf16)x;
    lo = (__bf16)(x - (float)hi);
}

// ---------------------------------------------------------------------------
// Weight pre-pass: W[K][N] f32 -> Wt_hi/Wt_lo[N][K] bf16 (error-split).
// ---------------------------------------------------------------------------
__global__ __launch_bounds__(256)
void wconvert_kernel(const float* __restrict__ W, int K, int N,
                     __bf16* __restrict__ Wt_hi, __bf16* __restrict__ Wt_lo) {
    __shared__ float tile[32][33];
    const int tid = threadIdx.x;
    const int kB = blockIdx.y * 32, nB = blockIdx.x * 32;
#pragma unroll
    for (int i = 0; i < 4; ++i) {
        int idx = i * 256 + tid;
        int k = idx >> 5, n = idx & 31;
        tile[k][n] = W[(size_t)(kB + k) * N + (nB + n)];
    }
    __syncthreads();
#pragma unroll
    for (int i = 0; i < 4; ++i) {
        int idx = i * 256 + tid;
        int n = idx >> 5, k = idx & 31;
        __bf16 hi, lo;
        split_bf16(tile[k][n], hi, lo);
        size_t o = (size_t)(nB + n) * K + (kB + k);
        Wt_hi[o] = hi;
        Wt_lo[o] = lo;
    }
}

// ---------------------------------------------------------------------------
// Activation pre-pass (for the network input x only): f32 -> hi/lo planes.
// ---------------------------------------------------------------------------
__global__ __launch_bounds__(256)
void split_kernel(const float* __restrict__ src,
                  __bf16* __restrict__ hi, __bf16* __restrict__ lo) {
    size_t idx = (size_t)blockIdx.x * 256 + threadIdx.x;   // float4 index
    float4 v = reinterpret_cast<const float4*>(src)[idx];
    float vv[4] = {v.x, v.y, v.z, v.w};
    v4bf h4, l4;
#pragma unroll
    for (int e = 0; e < 4; ++e) { __bf16 h, l; split_bf16(vv[e], h, l); h4[e] = h; l4[e] = l; }
    reinterpret_cast<v4bf*>(hi)[idx] = h4;
    reinterpret_cast<v4bf*>(lo)[idx] = l4;
}

// ---------------------------------------------------------------------------
// bf16x3 split-precision WMMA GEMM:  C[M,N] = epi( A[M,K] * W[K,N] )
//
// Both operands arrive as pre-split bf16 hi/lo planes (A row-major [M][K],
// B transposed [N][K]). Global->LDS is pure async DMA
// (global_load_async_to_lds_b128), DOUBLE-BUFFERED: tile i+1 is issued into
// the other LDS buffer right after the barrier, overlapping with the WMMA
// work on tile i; s_wait_asynccnt at the top of the next iteration only sees
// loads issued a full compute-phase earlier. One barrier per K-step.
// LDS tiles are K-contiguous with 16B-aligned 8-elem chunks -> each WMMA
// fragment is exactly two ds_load_b128 (ISA 7.12.2 lane/K mapping).
// Block tile 128x128, 8 wave32 waves, each wave 32x64 (2x4 WMMA tiles),
// 24 v_wmma per K-step per wave (hi*hi + lo*hi + hi*lo, fp32 accumulate).
// ---------------------------------------------------------------------------
#define TILE_M 128
#define TILE_N 128
#define TILE_K 32
#define LDSP   40   // row pitch in bf16 elems (80B: 16B-aligned, conflict-skewed)

__device__ __forceinline__ unsigned lds_off32(const void* p) {
    // ISA 10.2: LDS address = low 32 bits of the generic (flat) LDS pointer.
    return (unsigned)(size_t)p;
}

__device__ __forceinline__ void async_b128(unsigned lds_byte_off, const __bf16* gaddr) {
    asm volatile("global_load_async_to_lds_b128 %0, %1, off"
                 :: "v"(lds_byte_off), "v"((unsigned long long)gaddr)
                 : "memory");
}

__device__ __forceinline__ v16bf load_frag(const __bf16* base, int r, int hiLane) {
    const v8bf x0 = *(const v8bf*)(base + r * LDSP + 8 * hiLane);        // K chunk 0/1
    const v8bf x1 = *(const v8bf*)(base + r * LDSP + 16 + 8 * hiLane);   // K chunk 2/3
    return __builtin_shufflevector(x0, x1, 0, 1, 2, 3, 4, 5, 6, 7,
                                   8, 9, 10, 11, 12, 13, 14, 15);
}

template <int EPI, int WRITE_C, int WRITE_PL>
__global__ __launch_bounds__(256)
void gemm_bf16x3_wmma(const __bf16* __restrict__ Ahi, const __bf16* __restrict__ Alo, int lda,
                      const __bf16* __restrict__ Bhi, const __bf16* __restrict__ Blo,
                      float* __restrict__ C, int ldc,
                      __bf16* __restrict__ Phi, __bf16* __restrict__ Plo, int ldp,
                      int M, int N, int K,
                      const float* __restrict__ bias,
                      const float* __restrict__ Z, int ldz) {
    __shared__ alignas(16) __bf16 sAhi[2][TILE_M * LDSP];
    __shared__ alignas(16) __bf16 sAlo[2][TILE_M * LDSP];
    __shared__ alignas(16) __bf16 sBhi[2][TILE_N * LDSP];   // [col][k]
    __shared__ alignas(16) __bf16 sBlo[2][TILE_N * LDSP];

    const int tid  = threadIdx.x;
    const int lane = tid & 31;
    const int wid  = tid >> 5;          // 0..7
    const int wm   = wid >> 1;          // 0..3 -> 32-row slab
    const int wn   = wid & 1;           // 0..1 -> 64-col slab
    const int hiLane = (lane >> 4) & 1;

    const int mBase = blockIdx.y * TILE_M;
    const int nBase = blockIdx.x * TILE_N;

    const unsigned aHiB[2] = {lds_off32(sAhi[0]), lds_off32(sAhi[1])};
    const unsigned aLoB[2] = {lds_off32(sAlo[0]), lds_off32(sAlo[1])};
    const unsigned bHiB[2] = {lds_off32(sBhi[0]), lds_off32(sBhi[1])};
    const unsigned bLoB[2] = {lds_off32(sBlo[0]), lds_off32(sBlo[1])};

    // tail tile: pre-zero both B buffers once; OOB lanes never async-write
    if (nBase + TILE_N > N) {
        const __bf16 z = (__bf16)0.0f;
        v8bf zv = z;
#pragma unroll
        for (int i = 0; i < 10; ++i) {
            int idx = i * 256 + tid;            // 2560 chunks: 2 bufs x 2 planes x 640
            int buf = idx / 1280, rem = idx % 1280;
            __bf16* dst = (rem < 640) ? sBhi[buf] : sBlo[buf];
            *(v8bf*)(dst + (rem % 640) * 8) = zv;
        }
    }

    v8f acc[2][4];
#pragma unroll
    for (int i = 0; i < 2; ++i)
#pragma unroll
        for (int j = 0; j < 4; ++j) acc[i][j] = (v8f)0.0f;

    // per-thread tile slots: 512 16B chunks per plane, 2 per thread
    const int r0 = tid >> 2, c8 = (tid & 3) * 8;
    const int r1 = (256 + tid) >> 2;
    const unsigned u0 = (unsigned)(r0 * LDSP + c8) * 2u;
    const unsigned u1 = (unsigned)(r1 * LDSP + c8) * 2u;

    // running per-thread global pointers (advanced by TILE_K per issued tile)
    const __bf16* pAhi0 = Ahi + (size_t)(mBase + r0) * lda + c8;
    const __bf16* pAhi1 = Ahi + (size_t)(mBase + r1) * lda + c8;
    const __bf16* pAlo0 = Alo + (size_t)(mBase + r0) * lda + c8;
    const __bf16* pAlo1 = Alo + (size_t)(mBase + r1) * lda + c8;
    const bool gok0 = (nBase + r0) < N, gok1 = (nBase + r1) < N;
    const __bf16* pBhi0 = Bhi + (size_t)(nBase + r0) * K + c8;
    const __bf16* pBhi1 = Bhi + (size_t)(nBase + r1) * K + c8;
    const __bf16* pBlo0 = Blo + (size_t)(nBase + r0) * K + c8;
    const __bf16* pBlo1 = Blo + (size_t)(nBase + r1) * K + c8;

    auto issue = [&](int buf) {
        async_b128(aHiB[buf] + u0, pAhi0);  pAhi0 += TILE_K;
        async_b128(aHiB[buf] + u1, pAhi1);  pAhi1 += TILE_K;
        async_b128(aLoB[buf] + u0, pAlo0);  pAlo0 += TILE_K;
        async_b128(aLoB[buf] + u1, pAlo1);  pAlo1 += TILE_K;
        if (gok0) {
            async_b128(bHiB[buf] + u0, pBhi0);
            async_b128(bLoB[buf] + u0, pBlo0);
        }
        pBhi0 += TILE_K; pBlo0 += TILE_K;
        if (gok1) {
            async_b128(bHiB[buf] + u1, pBhi1);
            async_b128(bLoB[buf] + u1, pBlo1);
        }
        pBhi1 += TILE_K; pBlo1 += TILE_K;
    };

    __syncthreads();                 // order tail-zeroing before first issue
    issue(0);                        // prologue: tile 0 in flight

    const int T = K / TILE_K;
    for (int i = 0; i < T; ++i) {
        const int cur = i & 1;
        asm volatile("s_wait_asynccnt 0" ::: "memory");  // my tile-i writes landed
        __syncthreads();   // everyone's tile-i data visible; everyone's reads of
                           // the other buffer (tile i-1) retired (dscnt waited)
        if (i + 1 < T) issue(cur ^ 1);   // DMA of tile i+1 overlaps compute

        v16bf ah[2], al[2], bh[4], bl[4];
#pragma unroll
        for (int mi = 0; mi < 2; ++mi) {
            int r = wm * 32 + mi * 16 + (lane & 15);
            ah[mi] = load_frag(sAhi[cur], r, hiLane);
            al[mi] = load_frag(sAlo[cur], r, hiLane);
        }
#pragma unroll
        for (int ni = 0; ni < 4; ++ni) {
            int c = wn * 64 + ni * 16 + (lane & 15);
            bh[ni] = load_frag(sBhi[cur], c, hiLane);
            bl[ni] = load_frag(sBlo[cur], c, hiLane);
        }
#pragma unroll
        for (int mi = 0; mi < 2; ++mi) {
#pragma unroll
            for (int ni = 0; ni < 4; ++ni) {
                acc[mi][ni] = __builtin_amdgcn_wmma_f32_16x16x32_bf16(
                    false, ah[mi], false, bh[ni], (short)0, acc[mi][ni], false, false);
                acc[mi][ni] = __builtin_amdgcn_wmma_f32_16x16x32_bf16(
                    false, al[mi], false, bh[ni], (short)0, acc[mi][ni], false, false);
                acc[mi][ni] = __builtin_amdgcn_wmma_f32_16x16x32_bf16(
                    false, ah[mi], false, bl[ni], (short)0, acc[mi][ni], false, false);
            }
        }
    }

    // --- epilogue + stores (C/D layout: VGPR i -> M=i / M=i+8 by lane half) ---
    const int rOffLane = hiLane * 8;
    const int cLane    = lane & 15;
#pragma unroll
    for (int mi = 0; mi < 2; ++mi) {
#pragma unroll
        for (int ni = 0; ni < 4; ++ni) {
#pragma unroll
            for (int i = 0; i < 8; ++i) {
                int row = mBase + wm * 32 + mi * 16 + i + rOffLane;
                int col = nBase + wn * 64 + ni * 16 + cLane;
                if (col < N) {
                    float v = acc[mi][ni][i];
                    if (EPI == EPI_BIAS)           v += bias[col];
                    if (EPI == EPI_BIAS_SOFTPLUS)  v  = softplusf_(v + bias[col]);
                    if (EPI == EPI_SIGMUL)         v *= sigmoidf_(Z[(size_t)row * ldz + col]);
                    if (WRITE_C)  C[(size_t)row * ldc + col] = v;
                    if (WRITE_PL) {
                        __bf16 h, l;
                        split_bf16(v, h, l);
                        Phi[(size_t)row * ldp + col] = h;
                        Plo[(size_t)row * ldp + col] = l;
                    }
                }
            }
        }
    }
}

// ---------------------------------------------------------------------------
// Causal depthwise conv (K=4) + SiLU; writes xc f32 AND its hi/lo planes.
// ---------------------------------------------------------------------------
__global__ __launch_bounds__(256)
void conv_silu_kernel(const float* __restrict__ xz2,
                      const float* __restrict__ conv_w,
                      const float* __restrict__ conv_b,
                      float* __restrict__ xc,
                      __bf16* __restrict__ xc_hi, __bf16* __restrict__ xc_lo) {
    int g = blockIdx.x * 256 + threadIdx.x;            // (row, d)
    int row = g >> 11;
    int d   = g & (D_INNER - 1);
    int t   = row & (SEQL - 1);
    float4 w = *reinterpret_cast<const float4*>(conv_w + d * 4);
    const float* xm = xz2 + d;
    float acc = conv_b[d];
    float wv[4] = {w.x, w.y, w.z, w.w};
#pragma unroll
    for (int j = 0; j < D_CONVK; ++j) {
        int tt = t + j - (D_CONVK - 1);
        if (tt >= 0)
            acc += wv[j] * xm[(size_t)(row + j - (D_CONVK - 1)) * (2 * D_INNER)];
    }
    float v = acc * sigmoidf_(acc);
    size_t o = (size_t)row * D_INNER + d;
    xc[o] = v;
    __bf16 h, l;
    split_bf16(v, h, l);
    xc_hi[o] = h;
    xc_lo[o] = l;
}

// ---------------------------------------------------------------------------
// Selective scan: one thread per (b,d); h[16] in registers; serial over t.
// Fuses y = (scan + D_skip*u) * silu(zm); emits ys directly as hi/lo planes.
// ---------------------------------------------------------------------------
__global__ __launch_bounds__(256)
void selective_scan_kernel(const float* __restrict__ delta,
                           const float* __restrict__ xc,
                           const float* __restrict__ dbl,
                           const float* __restrict__ xz2,
                           const float* __restrict__ A_log,
                           const float* __restrict__ D_skip,
                           __bf16* __restrict__ ys_hi, __bf16* __restrict__ ys_lo) {
    int g = blockIdx.x * 256 + threadIdx.x;   // 0..8191
    int b = g >> 11;
    int d = g & (D_INNER - 1);

    float Aval[D_STATE];
#pragma unroll
    for (int n = 0; n < D_STATE; ++n)
        Aval[n] = -__expf(A_log[d * D_STATE + n]);
    const float Dk = D_skip[d];

    float h[D_STATE];
#pragma unroll
    for (int n = 0; n < D_STATE; ++n) h[n] = 0.0f;

    for (int t = 0; t < SEQL; ++t) {
        size_t row = (size_t)b * SEQL + t;
        float dlt = delta[row * D_INNER + d];
        float u   = xc[row * D_INNER + d];
        float zmv = xz2[row * (2 * D_INNER) + D_INNER + d];
        const float4* bc = reinterpret_cast<const float4*>(
            dbl + row * (DT_RANK + 2 * D_STATE) + DT_RANK);
        float Bv[D_STATE], Cv[D_STATE];
#pragma unroll
        for (int q = 0; q < 4; ++q) {
            float4 vB = bc[q], vC = bc[4 + q];
            Bv[q * 4 + 0] = vB.x; Bv[q * 4 + 1] = vB.y; Bv[q * 4 + 2] = vB.z; Bv[q * 4 + 3] = vB.w;
            Cv[q * 4 + 0] = vC.x; Cv[q * 4 + 1] = vC.y; Cv[q * 4 + 2] = vC.z; Cv[q * 4 + 3] = vC.w;
        }
        float du = dlt * u;
        float y = 0.0f;
#pragma unroll
        for (int n = 0; n < D_STATE; ++n) {
            h[n] = __expf(dlt * Aval[n]) * h[n] + du * Bv[n];
            y   += h[n] * Cv[n];
        }
        float v = (y + Dk * u) * (zmv * sigmoidf_(zmv));
        __bf16 vh, vl;
        split_bf16(v, vh, vl);
        ys_hi[row * D_INNER + d] = vh;
        ys_lo[row * D_INNER + d] = vl;
    }
}

// ---------------------------------------------------------------------------
// Launch: weight/activation pre-splits, 6 async-WMMA GEMMs, conv, scan.
// ---------------------------------------------------------------------------
extern "C" void kernel_launch(void* const* d_in, const int* in_sizes, int n_in,
                              void* d_out, int out_size, void* d_ws, size_t ws_size,
                              hipStream_t stream) {
    const float* x      = (const float*)d_in[0];
    const float* W_in_o = (const float*)d_in[1];
    const float* b_in_o = (const float*)d_in[2];
    const float* W_out_o= (const float*)d_in[3];
    const float* b_out_o= (const float*)d_in[4];
    const float* W_in_i = (const float*)d_in[5];
    const float* conv_w = (const float*)d_in[6];
    const float* conv_b = (const float*)d_in[7];
    const float* W_xp   = (const float*)d_in[8];
    const float* W_dt   = (const float*)d_in[9];
    const float* b_dt   = (const float*)d_in[10];
    const float* A_log  = (const float*)d_in[11];
    const float* D_skip = (const float*)d_in[12];
    const float* W_out_i= (const float*)d_in[13];
    float* out = (float*)d_out;

    // ---- workspace: f32 region, then bf16 plane region ----
    float* ws = (float*)d_ws;
    const size_t SZ_XZ  = (size_t)MROWS * 2 * D_INNER;
    const size_t SZ_XZ2 = (size_t)MROWS * 2 * D_INNER;
    const size_t SZ_XC  = (size_t)MROWS * D_INNER;
    const size_t SZ_DBL = (size_t)MROWS * (DT_RANK + 2 * D_STATE);
    const size_t SZ_DLT = (size_t)MROWS * D_INNER;
    float* ws_xz    = ws;
    float* ws_xz2   = ws_xz  + SZ_XZ;
    float* ws_xc    = ws_xz2 + SZ_XZ2;
    float* ws_dbl   = ws_xc  + SZ_XC;
    float* ws_delta = ws_dbl + SZ_DBL;
    float* f32_end  = ws_delta + SZ_DLT;

    __bf16* bp = (__bf16*)f32_end;
    const size_t WT   = (size_t)2048 * 4096;           // max weight plane
    const size_t PX   = (size_t)MROWS * D_MODEL;       // x plane
    const size_t PXZ  = (size_t)MROWS * 2 * D_INNER;   // xz plane
    const size_t PD   = (size_t)MROWS * D_INNER;       // d_inner-wide plane
    __bf16* wt_hi = bp;                 __bf16* wt_lo = wt_hi + WT;
    __bf16* x_hi  = wt_lo + WT;         __bf16* x_lo  = x_hi + PX;   // dead after s1
    __bf16* xz_hi = x_lo + PX;          __bf16* xz_lo = xz_hi + PXZ; // dead after s2
    __bf16* xc_hi = xz_lo + PXZ;        __bf16* xc_lo = xc_hi + PD;
    // region reuse (disjoint lifetimes):
    __bf16* dbl_hi = x_hi;              __bf16* dbl_lo = x_lo;       // in x region
    __bf16* ys_hi  = xz_hi;             __bf16* ys_lo  = xz_hi + PD; // in xz_hi region
    __bf16* o2_hi  = xz_lo;             __bf16* o2_lo  = xz_lo + PD; // in xz_lo region

    const dim3 blk(256);
    auto ggrid = [](int M, int N) { return dim3((N + TILE_N - 1) / TILE_N, M / TILE_M); };
    auto wgrid = [](int K, int N) { return dim3(N / 32, K / 32); };

    // 0) split network input x -> bf16 planes
    split_kernel<<<(MROWS * D_MODEL) / 1024, blk, 0, stream>>>(x, x_hi, x_lo);

    // 1) xz = x @ W_in_o + b_in_o            [8192,4096]   (f32 + planes)
    wconvert_kernel<<<wgrid(D_MODEL, 2 * D_INNER), blk, 0, stream>>>(
        W_in_o, D_MODEL, 2 * D_INNER, wt_hi, wt_lo);
    gemm_bf16x3_wmma<EPI_BIAS, 1, 1><<<ggrid(MROWS, 2 * D_INNER), blk, 0, stream>>>(
        x_hi, x_lo, D_MODEL, wt_hi, wt_lo, ws_xz, 2 * D_INNER,
        xz_hi, xz_lo, 2 * D_INNER, MROWS, 2 * D_INNER, D_MODEL, b_in_o, nullptr, 0);

    // 2) xz2 = x_proj @ W_in_i               [8192,4096]   (f32 only)
    wconvert_kernel<<<wgrid(D_INNER, 2 * D_INNER), blk, 0, stream>>>(
        W_in_i, D_INNER, 2 * D_INNER, wt_hi, wt_lo);
    gemm_bf16x3_wmma<EPI_NONE, 1, 0><<<ggrid(MROWS, 2 * D_INNER), blk, 0, stream>>>(
        xz_hi, xz_lo, 2 * D_INNER, wt_hi, wt_lo, ws_xz2, 2 * D_INNER,
        nullptr, nullptr, 0, MROWS, 2 * D_INNER, D_INNER, nullptr, nullptr, 0);

    // 3) xc = silu(depthwise_conv(xm) + b)   [8192,2048]   (f32 + planes)
    conv_silu_kernel<<<(MROWS * D_INNER) / 256, blk, 0, stream>>>(
        ws_xz2, conv_w, conv_b, ws_xc, xc_hi, xc_lo);

    // 4) dbl = xc @ W_xp                     [8192,160]    (f32 + planes)
    wconvert_kernel<<<wgrid(D_INNER, DT_RANK + 2 * D_STATE), blk, 0, stream>>>(
        W_xp, D_INNER, DT_RANK + 2 * D_STATE, wt_hi, wt_lo);
    gemm_bf16x3_wmma<EPI_NONE, 1, 1><<<ggrid(MROWS, DT_RANK + 2 * D_STATE), blk, 0, stream>>>(
        xc_hi, xc_lo, D_INNER, wt_hi, wt_lo, ws_dbl, DT_RANK + 2 * D_STATE,
        dbl_hi, dbl_lo, DT_RANK + 2 * D_STATE,
        MROWS, DT_RANK + 2 * D_STATE, D_INNER, nullptr, nullptr, 0);

    // 5) delta = softplus(dt @ W_dt + b_dt)  [8192,2048]   (dt = dbl[:, :128])
    wconvert_kernel<<<wgrid(DT_RANK, D_INNER), blk, 0, stream>>>(
        W_dt, DT_RANK, D_INNER, wt_hi, wt_lo);
    gemm_bf16x3_wmma<EPI_BIAS_SOFTPLUS, 1, 0><<<ggrid(MROWS, D_INNER), blk, 0, stream>>>(
        dbl_hi, dbl_lo, DT_RANK + 2 * D_STATE, wt_hi, wt_lo, ws_delta, D_INNER,
        nullptr, nullptr, 0, MROWS, D_INNER, DT_RANK, b_dt, nullptr, 0);

    // 6) ys = (scan + D*xc) * silu(zm)       [8192,2048]   (planes only)
    selective_scan_kernel<<<(BATCH * D_INNER) / 256, blk, 0, stream>>>(
        ws_delta, ws_xc, ws_dbl, ws_xz2, A_log, D_skip, ys_hi, ys_lo);

    // 7) out2 = (ys @ W_out_i) * sigmoid(z_proj)           (planes only)
    wconvert_kernel<<<wgrid(D_INNER, D_INNER), blk, 0, stream>>>(
        W_out_i, D_INNER, D_INNER, wt_hi, wt_lo);
    gemm_bf16x3_wmma<EPI_SIGMUL, 0, 1><<<ggrid(MROWS, D_INNER), blk, 0, stream>>>(
        ys_hi, ys_lo, D_INNER, wt_hi, wt_lo, nullptr, 0,
        o2_hi, o2_lo, D_INNER, MROWS, D_INNER, D_INNER, nullptr,
        ws_xz + D_INNER, 2 * D_INNER);

    // 8) out = out2 @ W_out_o + b_out_o      [8192,1024]   -> d_out
    wconvert_kernel<<<wgrid(D_INNER, D_MODEL), blk, 0, stream>>>(
        W_out_o, D_INNER, D_MODEL, wt_hi, wt_lo);
    gemm_bf16x3_wmma<EPI_BIAS, 1, 0><<<ggrid(MROWS, D_MODEL), blk, 0, stream>>>(
        o2_hi, o2_lo, D_INNER, wt_hi, wt_lo, out, D_MODEL,
        nullptr, nullptr, 0, MROWS, D_MODEL, D_INNER, b_out_o, nullptr, 0);
}
